// StandardController_77068893160245
// MI455X (gfx1250) — matrix operand order repaired and checked by reading, not compiled, not run on hardware
//
#include <hip/hip_runtime.h>
#include <hip/hip_bf16.h>

#define H_ 64
#define L_ 256
#define SLOTS_ 8
#define VOCAB_ 64
#define EPS_ 1e-5f

typedef __attribute__((ext_vector_type(16))) _Float16 v16h;
typedef __attribute__((ext_vector_type(8)))  _Float16 v8h;
typedef __attribute__((ext_vector_type(8)))  float    v8f;

// ---------------- LDS layout (bytes) ----------------
#define OFF_H    0          // float [256][64]   residual stream
#define OFF_A16  65536      // f16  [256][64]    h-f16 (QKV A) / P (softmax probs)
#define OFF_Q16  98304      // f16  [256][64]    Q   (FF16[256][128] aliases Q16+K16)
#define OFF_K16  131072     // f16  [256][64]    K
#define OFF_VT   163840     // f16  [64][256]    V transposed (feat-major)
#define OFF_TMP  196608     // float[256][64]    scores(64x256)/GEMM accum
#define OFF_ATT  262144     // f16  [256][64]    attention output
#define OFF_MISC 294912     // small reductions
#define SMEM_BYTES 303104

// Assert wave-uniformity of a tile coordinate so tile decomposition and
// region branches lower to SALU (s_cmp/s_cbranch) instead of EXEC masking.
__device__ __forceinline__ int uni(int x) {
  return __builtin_amdgcn_readfirstlane(x);
}

__device__ __forceinline__ v8f wmma16(v16h a, v16h b, v8f c) {
  return __builtin_amdgcn_wmma_f32_16x16x32_f16(false, a, false, b, (short)0, c,
                                                false, false);
}

// A fragment: 16x32 f16, row M = m0 + (lane&15); per lane two contiguous 8-half
// groups at K = k0 + (lane>>4)*16 and +8.  (2x ds_load_b128)
__device__ __forceinline__ v16h load_a16(const _Float16* base, int ldm, int m0,
                                         int k0) {
  const int lane = threadIdx.x & 31;
  const _Float16* p = base + (m0 + (lane & 15)) * ldm + k0 + (lane >> 4) * 16;
  v8h x = *(const v8h*)p;
  v8h y = *(const v8h*)(p + 8);
  v16h a;
#pragma unroll
  for (int i = 0; i < 8; ++i) { a[i] = x[i]; a[i + 8] = y[i]; }
  return a;
}

// B fragment: 32x16 f16 from a buffer laid out Bt[n][k] (column n contiguous
// in k).  Lane owns column n0+(lane&15), rows k0+(lane>>4)*16 .. +15.
__device__ __forceinline__ v16h load_bt16(const _Float16* base, int ldk, int n0,
                                          int k0) {
  const int lane = threadIdx.x & 31;
  const _Float16* p = base + (n0 + (lane & 15)) * ldk + k0 + (lane >> 4) * 16;
  v8h x = *(const v8h*)p;
  v8h y = *(const v8h*)(p + 8);
  v16h b;
#pragma unroll
  for (int i = 0; i < 8; ++i) { b[i] = x[i]; b[i + 8] = y[i]; }
  return b;
}

// B fragment from an f32 weight matrix W[N][K] in global memory (L2-resident),
// converted to f16 in registers.  16 contiguous floats per lane (4x b128).
__device__ __forceinline__ v16h load_bw32(const float* W, int ldk, int n0,
                                          int k0) {
  const int lane = threadIdx.x & 31;
  const float* p = W + (n0 + (lane & 15)) * ldk + k0 + (lane >> 4) * 16;
  v16h b;
#pragma unroll
  for (int t = 0; t < 4; ++t) {
    float4 f = ((const float4*)p)[t];
    b[4 * t + 0] = (_Float16)f.x;
    b[4 * t + 1] = (_Float16)f.y;
    b[4 * t + 2] = (_Float16)f.z;
    b[4 * t + 3] = (_Float16)f.w;
  }
  return b;
}

__global__ void zero_kernel(float* p, int n) {
  int i = blockIdx.x * blockDim.x + threadIdx.x;
  if (i < n) p[i] = 0.0f;
}

__global__ __launch_bounds__(256, 1)
void memnet_kernel(const int* __restrict__ seq, const int* __restrict__ query,
                   const int* __restrict__ target,
                   const float* __restrict__ embed_w,
                   const float* __restrict__ attn_in_w,
                   const float* __restrict__ attn_in_b,
                   const float* __restrict__ attn_out_w,
                   const float* __restrict__ attn_out_b,
                   const float* __restrict__ ff_w1,
                   const float* __restrict__ ff_b1,
                   const float* __restrict__ ff_w2,
                   const float* __restrict__ ff_b2,
                   const float* __restrict__ ln1_g,
                   const float* __restrict__ ln1_b,
                   const float* __restrict__ ln2_g,
                   const float* __restrict__ ln2_b,
                   const float* __restrict__ gate_w,
                   const float* __restrict__ gate_b,
                   const float* __restrict__ rq_w,
                   const float* __restrict__ rq_b,
                   const float* __restrict__ ro_w,
                   const float* __restrict__ ro_b,
                   const float* __restrict__ qe_w,
                   float* __restrict__ out, int B) {
  extern __shared__ char smem[];
  float*    hF    = (float*)(smem + OFF_H);      // [256][64]
  _Float16* A16   = (_Float16*)(smem + OFF_A16); // [256][64]
  _Float16* P16   = A16;                         // [64][256] during attention
  _Float16* Q16   = (_Float16*)(smem + OFF_Q16); // [256][64]
  _Float16* K16   = (_Float16*)(smem + OFF_K16); // [256][64]
  _Float16* FF16  = Q16;                         // [256][128] during FF
  _Float16* VT16  = (_Float16*)(smem + OFF_VT);  // [64][256]
  float*    TMPf  = (float*)(smem + OFF_TMP);    // [256][64] / scores [64][256]
  _Float16* ATT16 = (_Float16*)(smem + OFF_ATT); // [256][64]
  float*    miscF = (float*)(smem + OFF_MISC);
  float* red   = miscF + 0;    // [64][4]
  float* rsInv = miscF + 256;  // [64]   1/rowsum for softmax
  float* G     = miscF + 320;  // [256]  gate scores
  float* RQ    = miscF + 576;  // [64]
  float* RD    = miscF + 640;  // [64]
  float* LG    = miscF + 704;  // [64]
  float* RS    = miscF + 768;  // [8]
  float* RW    = miscF + 776;  // [8]
  int*   TOPK  = (int*)(miscF + 784); // [8]

  const int tid  = threadIdx.x;
  const int wave = uni(tid >> 5);     // wave index is uniform -> SGPR
  const int lane = tid & 31;
  const int laneLo = lane & 15;
  const int laneHi = lane >> 4;
  const int b = blockIdx.x;

  // Warm L2 with the (shared) weight matrices.
  if (tid == 0) {
    __builtin_prefetch(attn_in_w, 0, 1);
    __builtin_prefetch(ff_w1, 0, 1);
    __builtin_prefetch(ff_w2, 0, 1);
  }

  // ---- S0: token embedding -> hF (f32) and A16 (f16) ----
  {
    const int l = tid;  // L_ == blockDim.x == 256
    const int tok = seq[b * L_ + l];
    const float4* src = (const float4*)(embed_w + tok * H_);
    float* hr = hF + l * H_;
    _Float16* ar = A16 + l * H_;
#pragma unroll
    for (int i = 0; i < 16; ++i) {
      float4 f = src[i];
      hr[4 * i + 0] = f.x; hr[4 * i + 1] = f.y;
      hr[4 * i + 2] = f.z; hr[4 * i + 3] = f.w;
      ar[4 * i + 0] = (_Float16)f.x; ar[4 * i + 1] = (_Float16)f.y;
      ar[4 * i + 2] = (_Float16)f.z; ar[4 * i + 3] = (_Float16)f.w;
    }
  }
  __syncthreads();

  // ---- S1: QKV = h @ attn_in_w^T + b   (M=256, N=192, K=64) ----
  // Output tiles are 16 columns wide and n0 is a multiple of 16, so each tile
  // lands entirely in Q (n0<64), K (n0<128) or V (else); n0 is wave-uniform,
  // so the region split is a scalar branch.
  for (int t = wave; t < 16 * 12; t += 8) {
    const int m0 = uni((t & 15) * 16);
    const int n0 = uni((t >> 4) * 16);
    v8f c = {};
#pragma unroll
    for (int k0 = 0; k0 < H_; k0 += 32) {
      v16h a = load_a16(A16, H_, m0, k0);
      v16h w = load_bw32(attn_in_w, H_, n0, k0);
      c = wmma16(a, w, c);
    }
    const int n = n0 + laneLo;
    const float bias = attn_in_b[n];
    const int mBase = m0 + laneHi * 8;
    if (n0 < 64) {
      _Float16* q = Q16 + mBase * H_ + n;
#pragma unroll
      for (int r = 0; r < 8; ++r) q[r * H_] = (_Float16)(c[r] + bias);
    } else if (n0 < 128) {
      _Float16* k = K16 + mBase * H_ + (n - 64);
#pragma unroll
      for (int r = 0; r < 8; ++r) k[r * H_] = (_Float16)(c[r] + bias);
    } else {
      _Float16* v = VT16 + (n - 128) * L_ + mBase;  // transposed, unit stride
#pragma unroll
      for (int r = 0; r < 8; ++r) v[r] = (_Float16)(c[r] + bias);
    }
  }
  __syncthreads();

  // ---- S2: attention, per head, 64-query-row slabs ----
  const float sscale = 0.17677669529663687f;  // 1/sqrt(32)
  for (int head = 0; head < 2; ++head) {
    for (int qb = 0; qb < 4; ++qb) {
      // scores slab S[64][256] = Q_slab . K^T  (K=32)
      for (int t = wave; t < 4 * 16; t += 8) {
        const int m0 = uni((t & 3) * 16);   // row within slab
        const int n0 = uni((t >> 2) * 16);  // key tile
        v16h a = load_a16(Q16, H_, qb * 64 + m0, head * 32);
        v16h k = load_bt16(K16, H_, n0, head * 32);
        v8f c = {};
        c = wmma16(a, k, c);
#pragma unroll
        for (int r = 0; r < 8; ++r)
          TMPf[(m0 + r + laneHi * 8) * L_ + n0 + laneLo] = c[r] * sscale;
      }
      __syncthreads();

      // softmax over each of the 64 rows; 4 threads per row
      const int row  = tid >> 2;
      const int part = tid & 3;
      const float* srow = TMPf + row * L_ + part * 64;
      float pmax = -1e30f;
#pragma unroll 8
      for (int j = 0; j < 64; ++j) pmax = fmaxf(pmax, srow[j]);
      red[row * 4 + part] = pmax;
      __syncthreads();
      float rmax = fmaxf(fmaxf(red[row * 4 + 0], red[row * 4 + 1]),
                         fmaxf(red[row * 4 + 2], red[row * 4 + 3]));
      __syncthreads();
      float psum = 0.0f;
      _Float16* prow = P16 + row * L_ + part * 64;
#pragma unroll 8
      for (int j = 0; j < 64; ++j) {
        float e = __expf(srow[j] - rmax);
        prow[j] = (_Float16)e;   // unnormalized; fold 1/sum into O store
        psum += e;
      }
      red[row * 4 + part] = psum;
      __syncthreads();
      float rsum = red[row * 4 + 0] + red[row * 4 + 1] +
                   red[row * 4 + 2] + red[row * 4 + 3];
      if (part == 0) rsInv[row] = 1.0f / rsum;
      __syncthreads();

      // O = P . V   (M=64, N=32, K=256)  -> 8 tiles, one per wave
      {
        const int mt = uni(wave & 3);
        const int nt = uni(wave >> 2);
        v8f c = {};
#pragma unroll
        for (int k0 = 0; k0 < L_; k0 += 32) {
          v16h a = load_a16(P16, L_, mt * 16, k0);
          v16h v = load_bt16(VT16, L_, head * 32 + nt * 16, k0);
          c = wmma16(a, v, c);
        }
#pragma unroll
        for (int r = 0; r < 8; ++r) {
          const int mr = mt * 16 + r + laneHi * 8;  // row within slab
          ATT16[(qb * 64 + mr) * H_ + head * 32 + nt * 16 + laneLo] =
              (_Float16)(c[r] * rsInv[mr]);
        }
      }
      __syncthreads();
    }
  }

  // ---- S3: out-projection (M=256,N=64,K=64) + residual + LN1 ----
  for (int t = wave; t < 16 * 4; t += 8) {
    const int m0 = uni((t & 15) * 16);
    const int n0 = uni((t >> 4) * 16);
    v8f c = {};
#pragma unroll
    for (int k0 = 0; k0 < H_; k0 += 32) {
      v16h a = load_a16(ATT16, H_, m0, k0);
      v16h w = load_bw32(attn_out_w, H_, n0, k0);
      c = wmma16(a, w, c);
    }
    const int n = n0 + laneLo;
    const float bias = attn_out_b[n];
    float* d = TMPf + (m0 + laneHi * 8) * H_ + n;
#pragma unroll
    for (int r = 0; r < 8; ++r) d[r * H_] = c[r] + bias;
  }
  __syncthreads();
  {
    const int l = tid;
    float* hr = hF + l * H_;
    const float* tr = TMPf + l * H_;
    float mean = 0.0f;
#pragma unroll 8
    for (int j = 0; j < H_; ++j) mean += hr[j] + tr[j];
    mean *= (1.0f / H_);
    float var = 0.0f;
#pragma unroll 8
    for (int j = 0; j < H_; ++j) {
      float d = hr[j] + tr[j] - mean;
      var += d * d;
    }
    float inv = rsqrtf(var * (1.0f / H_) + EPS_);
#pragma unroll 8
    for (int j = 0; j < H_; ++j) {
      float y = (hr[j] + tr[j] - mean) * inv * ln1_g[j] + ln1_b[j];
      hr[j] = y;
      A16[l * H_ + j] = (_Float16)y;
    }
  }
  __syncthreads();

  // ---- S4: FF1 = relu(h @ ff_w1^T + b1)  (M=256,N=128,K=64) ----
  for (int t = wave; t < 16 * 8; t += 8) {
    const int m0 = uni((t & 15) * 16);
    const int n0 = uni((t >> 4) * 16);
    v8f c = {};
#pragma unroll
    for (int k0 = 0; k0 < H_; k0 += 32) {
      v16h a = load_a16(A16, H_, m0, k0);
      v16h w = load_bw32(ff_w1, H_, n0, k0);
      c = wmma16(a, w, c);
    }
    const int n = n0 + laneLo;
    const float bias = ff_b1[n];
    _Float16* d = FF16 + (m0 + laneHi * 8) * 128 + n;
#pragma unroll
    for (int r = 0; r < 8; ++r) {
      float v = c[r] + bias;
      v = v > 0.0f ? v : 0.0f;
      d[r * 128] = (_Float16)v;
    }
  }
  __syncthreads();

  // ---- S5: FF2 (M=256,N=64,K=128) + residual + LN2 + gate ----
  for (int t = wave; t < 16 * 4; t += 8) {
    const int m0 = uni((t & 15) * 16);
    const int n0 = uni((t >> 4) * 16);
    v8f c = {};
#pragma unroll
    for (int k0 = 0; k0 < 128; k0 += 32) {
      v16h a = load_a16(FF16, 128, m0, k0);
      v16h w = load_bw32(ff_w2, 128, n0, k0);
      c = wmma16(a, w, c);
    }
    const int n = n0 + laneLo;
    const float bias = ff_b2[n];
    float* d = TMPf + (m0 + laneHi * 8) * H_ + n;
#pragma unroll
    for (int r = 0; r < 8; ++r) d[r * H_] = c[r] + bias;
  }
  __syncthreads();
  {
    const int l = tid;
    float* hr = hF + l * H_;
    const float* tr = TMPf + l * H_;
    float mean = 0.0f;
#pragma unroll 8
    for (int j = 0; j < H_; ++j) mean += hr[j] + tr[j];
    mean *= (1.0f / H_);
    float var = 0.0f;
#pragma unroll 8
    for (int j = 0; j < H_; ++j) {
      float d = hr[j] + tr[j] - mean;
      var += d * d;
    }
    float inv = rsqrtf(var * (1.0f / H_) + EPS_);
    float gdot = 0.0f;
#pragma unroll 8
    for (int j = 0; j < H_; ++j) {
      float y = (hr[j] + tr[j] - mean) * inv * ln2_g[j] + ln2_b[j];
      hr[j] = y;
      gdot += y * gate_w[j];
    }
    G[l] = 1.0f / (1.0f + __expf(-(gdot + gate_b[0])));
  }
  __syncthreads();

  // ---- S6: top-8 gate scores (first-index tie-break, matches lax.top_k) ----
  if (tid == 0) {
#pragma unroll
    for (int s = 0; s < SLOTS_; ++s) {
      float best = -1e30f;
      int bi = 0;
      for (int i = 0; i < L_; ++i) {
        float v = G[i];
        if (v > best) { best = v; bi = i; }
      }
      TOPK[s] = bi;
      G[bi] = -1e30f;
    }
  }
  __syncthreads();

  // ---- S7: memory reader + cross-entropy ----
  if (tid < H_) {
    const int i = tid;
    const float* qh = qe_w + query[b] * H_;
    float acc = rq_b[i];
#pragma unroll 8
    for (int j = 0; j < H_; ++j) acc += rq_w[i * H_ + j] * qh[j];
    RQ[i] = acc;
  }
  __syncthreads();
  if (tid < SLOTS_) {
    const float* mrow = hF + TOPK[tid] * H_;
    float acc = 0.0f;
#pragma unroll 8
    for (int j = 0; j < H_; ++j) acc += RQ[j] * mrow[j];
    RS[tid] = acc * 0.125f;  // 1/sqrt(64)
  }
  __syncthreads();
  if (tid == 0) {
    float mx = -1e30f;
    for (int s = 0; s < SLOTS_; ++s) mx = fmaxf(mx, RS[s]);
    float sum = 0.0f;
    for (int s = 0; s < SLOTS_; ++s) { RW[s] = __expf(RS[s] - mx); sum += RW[s]; }
    float inv = 1.0f / sum;
    for (int s = 0; s < SLOTS_; ++s) RW[s] *= inv;
  }
  __syncthreads();
  if (tid < H_) {
    float acc = 0.0f;
#pragma unroll
    for (int s = 0; s < SLOTS_; ++s) acc += RW[s] * hF[TOPK[s] * H_ + tid];
    RD[tid] = acc;
  }
  __syncthreads();
  if (tid < VOCAB_) {
    float acc = ro_b[tid];
#pragma unroll 8
    for (int j = 0; j < H_; ++j) acc += RD[j] * ro_w[tid * H_ + j];
    LG[tid] = acc;
  }
  __syncthreads();
  if (tid == 0) {
    float mx = -1e30f;
    for (int v = 0; v < VOCAB_; ++v) mx = fmaxf(mx, LG[v]);
    float sum = 0.0f;
    for (int v = 0; v < VOCAB_; ++v) sum += __expf(LG[v] - mx);
    float lse = mx + __logf(sum);
    float nll = lse - LG[target[b]];
    atomicAdd(out, nll * (1.0f / (float)B));
  }
}

extern "C" void kernel_launch(void* const* d_in, const int* in_sizes, int n_in,
                              void* d_out, int out_size, void* d_ws,
                              size_t ws_size, hipStream_t stream) {
  (void)n_in; (void)d_ws; (void)ws_size;
  const int*   seq        = (const int*)d_in[0];
  const int*   query      = (const int*)d_in[1];
  const int*   target     = (const int*)d_in[2];
  const float* embed_w    = (const float*)d_in[3];
  const float* attn_in_w  = (const float*)d_in[4];
  const float* attn_in_b  = (const float*)d_in[5];
  const float* attn_out_w = (const float*)d_in[6];
  const float* attn_out_b = (const float*)d_in[7];
  const float* ff_w1      = (const float*)d_in[8];
  const float* ff_b1      = (const float*)d_in[9];
  const float* ff_w2      = (const float*)d_in[10];
  const float* ff_b2      = (const float*)d_in[11];
  const float* ln1_g      = (const float*)d_in[12];
  const float* ln1_b      = (const float*)d_in[13];
  const float* ln2_g      = (const float*)d_in[14];
  const float* ln2_b      = (const float*)d_in[15];
  const float* gate_w     = (const float*)d_in[16];
  const float* gate_b     = (const float*)d_in[17];
  const float* rq_w       = (const float*)d_in[18];
  const float* rq_b       = (const float*)d_in[19];
  const float* ro_w       = (const float*)d_in[20];
  const float* ro_b       = (const float*)d_in[21];
  const float* qe_w       = (const float*)d_in[22];

  const int B = in_sizes[1];  // query has shape (B,)
  zero_kernel<<<1, 32, 0, stream>>>((float*)d_out, out_size);
  memnet_kernel<<<B, 256, SMEM_BYTES, stream>>>(
      seq, query, target, embed_w, attn_in_w, attn_in_b, attn_out_w, attn_out_b,
      ff_w1, ff_b1, ff_w2, ff_b2, ln1_g, ln1_b, ln2_g, ln2_b, gate_w, gate_b,
      rq_w, rq_b, ro_w, ro_b, qe_w, (float*)d_out, B);
}
